// RWKV_Tmix_x070_Mod_42004780155477
// MI455X (gfx1250) — compile-verified, matrix operand order checked by reading, not run on hardware
//
#include <hip/hip_runtime.h>
#include <hip/hip_bf16.h>

// ---------------------------------------------------------------------------
// RWKV7 Tmix for MI455X (gfx1250, wave32, WMMA 16x16x32 bf16 + async LDS DMA)
// ---------------------------------------------------------------------------

typedef __attribute__((ext_vector_type(16))) __bf16 v16bf;
typedef __attribute__((ext_vector_type(8)))  float  v8f;
typedef int v4i_vec __attribute__((vector_size(16)));   // matches async builtin param

static constexpr int BB = 2, TT = 1024, CC = 1024, HH = 16, NN = 64;
static constexpr int MROWS = BB * TT;                 // 2048 tokens
static constexpr size_t MC = (size_t)MROWS * CC;      // 2,097,152

// ---- workspace layout (bytes) ----
static constexpr size_t SZ_MC_BF = MC * 2;            // 4 MiB
static constexpr size_t SZ_MC_F  = MC * 4;            // 8 MiB
static constexpr size_t SZ_CC_BF = (size_t)CC * CC * 2;

static constexpr size_t OFF_XR  = 0;
static constexpr size_t OFF_XW  = OFF_XR + SZ_MC_BF;
static constexpr size_t OFF_XK  = OFF_XW + SZ_MC_BF;
static constexpr size_t OFF_XV  = OFF_XK + SZ_MC_BF;
static constexpr size_t OFF_XA  = OFF_XV + SZ_MC_BF;
static constexpr size_t OFF_XG  = OFF_XA + SZ_MC_BF;
static constexpr size_t OFF_WTR = OFF_XG + SZ_MC_BF;
static constexpr size_t OFF_WTK = OFF_WTR + SZ_CC_BF;
static constexpr size_t OFF_WTV = OFF_WTK + SZ_CC_BF;
static constexpr size_t OFF_WTO = OFF_WTV + SZ_CC_BF;
static constexpr size_t OFF_W1T = OFF_WTO + SZ_CC_BF;             // 64x1024
static constexpr size_t OFF_W2T = OFF_W1T + 64 * 1024 * 2;        // 1024x64
static constexpr size_t OFF_A1T = OFF_W2T + 64 * 1024 * 2;
static constexpr size_t OFF_A2T = OFF_A1T + 64 * 1024 * 2;
static constexpr size_t OFF_V1T = OFF_A2T + 64 * 1024 * 2;        // 32x1024
static constexpr size_t OFF_V2T = OFF_V1T + 32 * 1024 * 2;
static constexpr size_t OFF_G1T = OFF_V2T + 32 * 1024 * 2;        // 128x1024
static constexpr size_t OFF_G2T = OFF_G1T + 128 * 1024 * 2;
static constexpr size_t OFF_HW  = OFF_G2T + 128 * 1024 * 2;       // M x 64 bf16
static constexpr size_t OFF_HA  = OFF_HW + (size_t)MROWS * 64 * 2;
static constexpr size_t OFF_HV  = OFF_HA + (size_t)MROWS * 64 * 2;    // M x 32
static constexpr size_t OFF_HG  = OFF_HV + (size_t)MROWS * 32 * 2;    // M x 128
static constexpr size_t OFF_YG  = OFF_HG + (size_t)MROWS * 128 * 2;   // M x C bf16
static constexpr size_t OFF_RF  = (OFF_YG + SZ_MC_BF + 255) & ~(size_t)255;
static constexpr size_t OFF_KF  = OFF_RF + SZ_MC_F;
static constexpr size_t OFF_VF  = OFF_KF + SZ_MC_F;
static constexpr size_t OFF_WD  = OFF_VF + SZ_MC_F;   // decay (in-place)
static constexpr size_t OFF_AB  = OFF_WD + SZ_MC_F;   // b = kk*a (in-place)
static constexpr size_t OFF_VLR = OFF_AB + SZ_MC_F;   // v gate low-rank out
static constexpr size_t OFF_KK  = OFF_VLR + SZ_MC_F;  // normalized kk
static constexpr size_t OFF_GF  = OFF_KK + SZ_MC_F;   // g
static constexpr size_t OFF_YF  = OFF_GF + SZ_MC_F;   // recurrence output

__device__ __forceinline__ float sigf(float x) { return 1.0f / (1.0f + __expf(-x)); }
__device__ __forceinline__ float wave_sum32(float v) {
  #pragma unroll
  for (int o = 16; o; o >>= 1) v += __shfl_xor(v, o);
  return v;
}

// ---- CDNA5 async global->LDS path (probe-guarded; sync fallback keeps the
//      same barrier structure correct on both paths) ----
#if defined(__HIP_DEVICE_COMPILE__) && \
    __has_builtin(__builtin_amdgcn_global_load_async_to_lds_b128) && \
    __has_builtin(__builtin_amdgcn_s_wait_asynccnt)
#define RWKV_ASYNC_LDS 1
#else
#define RWKV_ASYNC_LDS 0
#endif

__device__ __forceinline__ void cp16_to_lds(char* lds_dst, const __bf16* g_src) {
#if RWKV_ASYNC_LDS
  __builtin_amdgcn_global_load_async_to_lds_b128(
      (__attribute__((address_space(1))) v4i_vec*)g_src,
      (__attribute__((address_space(3))) v4i_vec*)lds_dst, 0, 0);
#else
  *(uint4*)lds_dst = *(const uint4*)g_src;
#endif
}
template <int N>
__device__ __forceinline__ void wait_async_le() {
#if RWKV_ASYNC_LDS
  __builtin_amdgcn_s_wait_asynccnt(N);   // immediate required
#endif
}

// ---------------------------------------------------------------------------
// fp32 (K x N) -> bf16 transposed (N x K): LDS-tiled coalesced transpose
// (all dims here are multiples of 32)
// ---------------------------------------------------------------------------
__global__ __launch_bounds__(256)
void conv_t_kernel(const float* __restrict__ src, __bf16* __restrict__ dst,
                   int K, int N) {
  __shared__ float tile[32][33];
  const int tx = threadIdx.x & 31;
  const int ty = threadIdx.x >> 5;       // 0..7
  const int n0 = blockIdx.x * 32;
  const int k0 = blockIdx.y * 32;
  #pragma unroll
  for (int j = 0; j < 4; ++j)
    tile[ty + 8 * j][tx] = src[(size_t)(k0 + ty + 8 * j) * N + n0 + tx];
  __syncthreads();
  #pragma unroll
  for (int j = 0; j < 4; ++j)
    dst[(size_t)(n0 + ty + 8 * j) * K + k0 + tx] = (__bf16)tile[tx][ty + 8 * j];
}

// ---------------------------------------------------------------------------
// token shift + six mixes -> bf16
// ---------------------------------------------------------------------------
__global__ void mix_kernel(const float* __restrict__ x,
                           const float* __restrict__ mr, const float* __restrict__ mw,
                           const float* __restrict__ mk, const float* __restrict__ mv,
                           const float* __restrict__ ma, const float* __restrict__ mg,
                           __bf16* __restrict__ xr, __bf16* __restrict__ xw,
                           __bf16* __restrict__ xk, __bf16* __restrict__ xv,
                           __bf16* __restrict__ xa, __bf16* __restrict__ xg) {
  size_t idx = (size_t)blockIdx.x * blockDim.x + threadIdx.x;
  if (idx >= MC) return;
  int c = (int)(idx % CC);
  size_t row = idx / CC;
  int t = (int)(row % TT);
  float cur = x[idx];
  float prev = (t > 0) ? x[idx - CC] : 0.0f;
  float xx = prev - cur;
  xr[idx] = (__bf16)(cur + xx * mr[c]);
  xw[idx] = (__bf16)(cur + xx * mw[c]);
  xk[idx] = (__bf16)(cur + xx * mk[c]);
  xv[idx] = (__bf16)(cur + xx * mv[c]);
  xa[idx] = (__bf16)(cur + xx * ma[c]);
  xg[idx] = (__bf16)(cur + xx * mg[c]);
}

// ---------------------------------------------------------------------------
// bf16 WMMA GEMM:  C[M x N] = A[M x K] * Bt[N x K]^T
// block = 4 waves (128 thr) -> 128x128 tile; each wave 64x64 via 16 WMMAs.
// Tiles staged in LDS (double buffered) via GLOBAL_LOAD_ASYNC_TO_LDS_B128:
// each thread moves 4x16B of A and 4x16B of B per K-step (exactly 8 async
// issues per wave -> s_wait_asynccnt 8 retires the in-flight tile, since
// async loads complete in order). OOB B rows are clamped (never stored).
// epilogue: 0 = store f32, 1 = tanh->bf16, 2 = sigmoid->bf16, 3 = plain bf16
// ---------------------------------------------------------------------------
union FragBF { uint4 u[2]; v16bf v; };

__global__ __launch_bounds__(128)
void gemm_bf16_wmma(const __bf16* __restrict__ A, const __bf16* __restrict__ Bt,
                    float* __restrict__ Cf, __bf16* __restrict__ Cb,
                    int Ndim, int Kdim, int epilogue) {
  __shared__ __align__(16) char smem[2][16384];   // [buf][ A(8K) | B(8K) ]
  const int tid  = threadIdx.x;
  const int lane = tid & 31;
  const int wave = tid >> 5;
  const int wm   = (wave >> 1) * 64;
  const int wn   = (wave & 1) * 64;
  const int bm   = blockIdx.x * 128;
  const int bn   = blockIdx.y * 128;
  const int lrow = lane & 15;
  const int khalf = lane >> 4;

  const int r0 = tid >> 2;     // staging: rows r0+{0,32,64,96}, 16B chunk c4
  const int c4 = tid & 3;

  auto issue = [&](int buf, int kb) {
    char* sA = smem[buf];
    char* sB = smem[buf] + 8192;
    #pragma unroll
    for (int j = 0; j < 4; ++j) {
      const int row = r0 + 32 * j;
      cp16_to_lds(sA + row * 64 + c4 * 16,
                  A + (size_t)(bm + row) * Kdim + kb + c4 * 8);
    }
    #pragma unroll
    for (int j = 0; j < 4; ++j) {
      const int row = r0 + 32 * j;
      const int nr = (bn + row < Ndim) ? (bn + row) : (Ndim - 1);  // clamp
      cp16_to_lds(sB + row * 64 + c4 * 16,
                  Bt + (size_t)nr * Kdim + kb + c4 * 8);
    }
  };

  v8f acc[4][4] = {};
  const int ntiles = Kdim >> 5;

  issue(0, 0);
  for (int it = 0; it < ntiles; ++it) {
    const int cur = it & 1;
    if (it + 1 < ntiles) {
      issue(cur ^ 1, (it + 1) << 5);
      wait_async_le<8>();        // oldest 8 (current tile) complete
    } else {
      wait_async_le<0>();
    }
    __syncthreads();             // all waves' tile data visible

    const char* sA = smem[cur];
    const char* sB = smem[cur] + 8192;
    FragBF af[4], bfr[4];
    #pragma unroll
    for (int mt = 0; mt < 4; ++mt) {
      const char* p = sA + (wm + mt * 16 + lrow) * 64 + khalf * 16;
      af[mt].u[0] = *(const uint4*)p;
      af[mt].u[1] = *(const uint4*)(p + 32);
    }
    #pragma unroll
    for (int nt = 0; nt < 4; ++nt) {
      const char* p = sB + (wn + nt * 16 + lrow) * 64 + khalf * 16;
      bfr[nt].u[0] = *(const uint4*)p;
      bfr[nt].u[1] = *(const uint4*)(p + 32);
    }
    #pragma unroll
    for (int mt = 0; mt < 4; ++mt)
      #pragma unroll
      for (int nt = 0; nt < 4; ++nt)
        acc[mt][nt] = __builtin_amdgcn_wmma_f32_16x16x32_bf16(
            false, af[mt].v, false, bfr[nt].v, (short)0, acc[mt][nt], false, false);
    __syncthreads();             // reads done before buffer is overwritten
  }

  // C/D layout: lane<16 -> rows 0..7 (vgpr 0..7), col=lane; lane>=16 -> rows 8..15
  const int srow = khalf * 8;
  const int scol = lane & 15;
  #pragma unroll
  for (int mt = 0; mt < 4; ++mt) {
    #pragma unroll
    for (int nt = 0; nt < 4; ++nt) {
      const int n = bn + wn + nt * 16 + scol;
      if (n >= Ndim) continue;
      const int mbase = bm + wm + mt * 16 + srow;
      #pragma unroll
      for (int r = 0; r < 8; ++r) {
        float val = acc[mt][nt][r];
        size_t off = (size_t)(mbase + r) * Ndim + n;
        if (epilogue == 0) {
          Cf[off] = val;
        } else {
          if (epilogue == 1) val = tanhf(val);
          else if (epilogue == 2) val = sigf(val);
          Cb[off] = (__bf16)val;
        }
      }
    }
  }
}

// ---------------------------------------------------------------------------
// E1: per (token, head) — decay, a, v-gate, kk normalize, k rescale, b = kk*a
// ---------------------------------------------------------------------------
__global__ __launch_bounds__(32)
void e1_kernel(float* __restrict__ Kf, float* __restrict__ Vf,
               float* __restrict__ Wd, float* __restrict__ Ab,
               const float* __restrict__ Vlr, const float* __restrict__ vfirst,
               const float* __restrict__ w0, const float* __restrict__ a0,
               const float* __restrict__ v0, const float* __restrict__ k_k,
               const float* __restrict__ k_a, float* __restrict__ KKf) {
  const int tok = blockIdx.x, h = blockIdx.y, lane = threadIdx.x;
  const size_t base = (size_t)tok * CC + h * NN;
  const int c0 = h * NN + lane, c1 = c0 + 32;
  const size_t i0 = base + lane, i1 = i0 + 32;

  float k0 = Kf[i0], k1 = Kf[i1];
  float kk0 = k0 * k_k[c0], kk1 = k1 * k_k[c1];
  float ss = wave_sum32(kk0 * kk0 + kk1 * kk1);
  float inv = 1.0f / fmaxf(sqrtf(ss), 1e-12f);
  kk0 *= inv; kk1 *= inv;

  float av0 = sigf(Ab[i0] + a0[c0]);
  float av1 = sigf(Ab[i1] + a0[c1]);

  float kf0 = k0 * (1.0f + (av0 - 1.0f) * k_a[c0]);
  float kf1 = k1 * (1.0f + (av1 - 1.0f) * k_a[c1]);

  float wv0 = Wd[i0] + w0[c0], wv1 = Wd[i1] + w0[c1];
  float w_0 = -log1pf(__expf(-wv0)) - 0.5f;   // -softplus(-x) - 0.5
  float w_1 = -log1pf(__expf(-wv1)) - 0.5f;
  float d0 = __expf(-__expf(w_0)), d1 = __expf(-__expf(w_1));

  float vg0 = sigf(Vlr[i0] + v0[c0]), vg1 = sigf(Vlr[i1] + v0[c1]);
  float v_0 = Vf[i0], v_1 = Vf[i1];
  v_0 = v_0 + (vfirst[i0] - v_0) * vg0;
  v_1 = v_1 + (vfirst[i1] - v_1) * vg1;

  Kf[i0] = kf0;  Kf[i1] = kf1;
  Vf[i0] = v_0;  Vf[i1] = v_1;
  Wd[i0] = d0;   Wd[i1] = d1;
  KKf[i0] = kk0; KKf[i1] = kk1;
  Ab[i0] = kk0 * av0; Ab[i1] = kk1 * av1;    // b = kk * a
}

// ---------------------------------------------------------------------------
// Recurrence: S_t = S diag(d) + (S @ (-kk)) b^T + v k^T ; out = S @ r
// 32 blocks (b,h), 256 threads: 4 lanes per state row, 16 cols each in VGPRs.
// ---------------------------------------------------------------------------
__global__ __launch_bounds__(256)
void rwkv_rec_kernel(const float* __restrict__ Rf, const float* __restrict__ Dd,
                     const float* __restrict__ Kf, const float* __restrict__ Vf,
                     const float* __restrict__ KKf, const float* __restrict__ Bv,
                     float* __restrict__ Y) {
  const int bh = blockIdx.x;
  const int b = bh / HH, h = bh % HH;
  const int tid = threadIdx.x;
  const int i = tid >> 2;        // state row
  const int q = tid & 3;         // column quadrant
  const int j0 = q * 16;

  __shared__ float lv[6 * 64];   // [R, D, K, V, KK, B]

  float S[16];
  #pragma unroll
  for (int jj = 0; jj < 16; ++jj) S[jj] = 0.0f;

  const size_t headoff = (size_t)h * NN;
  for (int t = 0; t < TT; ++t) {
    const size_t rowoff = ((size_t)b * TT + t) * CC + headoff;
    for (int idx = tid; idx < 6 * 64; idx += 256) {
      const int vsel = idx >> 6, e = idx & 63;
      const float* src = (vsel == 0) ? Rf : (vsel == 1) ? Dd : (vsel == 2) ? Kf
                       : (vsel == 3) ? Vf : (vsel == 4) ? KKf : Bv;
      lv[idx] = src[rowoff + e];
    }
    __syncthreads();

    float sa = 0.0f;
    #pragma unroll
    for (int jj = 0; jj < 16; ++jj) sa += S[jj] * lv[4 * 64 + j0 + jj];
    sa += __shfl_xor(sa, 1); sa += __shfl_xor(sa, 2);
    sa = -sa;                                    // a_t = -kk

    const float vi = lv[3 * 64 + i];
    float out = 0.0f;
    #pragma unroll
    for (int jj = 0; jj < 16; ++jj) {
      const int j = j0 + jj;
      float s = S[jj] * lv[64 + j] + sa * lv[5 * 64 + j] + vi * lv[2 * 64 + j];
      S[jj] = s;
      out += s * lv[j];
    }
    out += __shfl_xor(out, 1); out += __shfl_xor(out, 2);
    if (q == 0) Y[rowoff + i] = out;
    __syncthreads();
  }
}

// ---------------------------------------------------------------------------
// E2: GroupNorm(H groups) + bonus + gate, emit bf16 for final GEMM
// ---------------------------------------------------------------------------
__global__ __launch_bounds__(32)
void e2_kernel(const float* __restrict__ Y, const float* __restrict__ Rf,
               const float* __restrict__ Kf, const float* __restrict__ Vf,
               const float* __restrict__ Gf, const float* __restrict__ r_k,
               const float* __restrict__ ln_w, const float* __restrict__ ln_b,
               __bf16* __restrict__ YG) {
  const int tok = blockIdx.x, h = blockIdx.y, lane = threadIdx.x;
  const size_t base = (size_t)tok * CC + h * NN;
  const int c0 = h * NN + lane, c1 = c0 + 32;
  const size_t i0 = base + lane, i1 = i0 + 32;

  float y0 = Y[i0], y1 = Y[i1];
  float mean = wave_sum32(y0 + y1) * (1.0f / 64.0f);
  float var  = wave_sum32(y0 * y0 + y1 * y1) * (1.0f / 64.0f) - mean * mean;
  float inv  = rsqrtf(var + 64e-5f);

  float bsum = wave_sum32(Rf[i0] * Kf[i0] * r_k[c0] + Rf[i1] * Kf[i1] * r_k[c1]);

  float o0 = ((y0 - mean) * inv) * ln_w[c0] + ln_b[c0] + bsum * Vf[i0];
  float o1 = ((y1 - mean) * inv) * ln_w[c1] + ln_b[c1] + bsum * Vf[i1];
  YG[i0] = (__bf16)(o0 * Gf[i0]);
  YG[i1] = (__bf16)(o1 * Gf[i1]);
}

__global__ void copy4_f32_kernel(const float* __restrict__ src, float* __restrict__ dst,
                                 size_t n4) {
  size_t idx = (size_t)blockIdx.x * blockDim.x + threadIdx.x;
  if (idx < n4) ((float4*)dst)[idx] = ((const float4*)src)[idx];
}

// ---------------------------------------------------------------------------
extern "C" void kernel_launch(void* const* d_in, const int* in_sizes, int n_in,
                              void* d_out, int out_size, void* d_ws, size_t ws_size,
                              hipStream_t stream) {
  const float* x      = (const float*)d_in[0];
  const float* vfirst = (const float*)d_in[1];
  const float* x_r = (const float*)d_in[2]; const float* x_w = (const float*)d_in[3];
  const float* x_k = (const float*)d_in[4]; const float* x_v = (const float*)d_in[5];
  const float* x_a = (const float*)d_in[6]; const float* x_g = (const float*)d_in[7];
  const float* w1 = (const float*)d_in[8];  const float* w2 = (const float*)d_in[9];
  const float* w0 = (const float*)d_in[10];
  const float* a1 = (const float*)d_in[11]; const float* a2 = (const float*)d_in[12];
  const float* a0 = (const float*)d_in[13];
  const float* v1 = (const float*)d_in[14]; const float* v2 = (const float*)d_in[15];
  const float* v0 = (const float*)d_in[16];
  const float* g1 = (const float*)d_in[17]; const float* g2 = (const float*)d_in[18];
  const float* k_k = (const float*)d_in[19]; const float* k_a = (const float*)d_in[20];
  const float* r_k = (const float*)d_in[21];
  const float* W_r = (const float*)d_in[22]; const float* W_k = (const float*)d_in[23];
  const float* W_v = (const float*)d_in[24]; const float* W_o = (const float*)d_in[25];
  const float* ln_w = (const float*)d_in[26]; const float* ln_b = (const float*)d_in[27];

  char* ws = (char*)d_ws;
  __bf16* XR = (__bf16*)(ws + OFF_XR);  __bf16* XW = (__bf16*)(ws + OFF_XW);
  __bf16* XK = (__bf16*)(ws + OFF_XK);  __bf16* XV = (__bf16*)(ws + OFF_XV);
  __bf16* XA = (__bf16*)(ws + OFF_XA);  __bf16* XG = (__bf16*)(ws + OFF_XG);
  __bf16* WTR = (__bf16*)(ws + OFF_WTR); __bf16* WTK = (__bf16*)(ws + OFF_WTK);
  __bf16* WTV = (__bf16*)(ws + OFF_WTV); __bf16* WTO = (__bf16*)(ws + OFF_WTO);
  __bf16* W1T = (__bf16*)(ws + OFF_W1T); __bf16* W2T = (__bf16*)(ws + OFF_W2T);
  __bf16* A1T = (__bf16*)(ws + OFF_A1T); __bf16* A2T = (__bf16*)(ws + OFF_A2T);
  __bf16* V1T = (__bf16*)(ws + OFF_V1T); __bf16* V2T = (__bf16*)(ws + OFF_V2T);
  __bf16* G1T = (__bf16*)(ws + OFF_G1T); __bf16* G2T = (__bf16*)(ws + OFF_G2T);
  __bf16* HW = (__bf16*)(ws + OFF_HW);  __bf16* HA = (__bf16*)(ws + OFF_HA);
  __bf16* HV = (__bf16*)(ws + OFF_HV);  __bf16* HG = (__bf16*)(ws + OFF_HG);
  __bf16* YG = (__bf16*)(ws + OFF_YG);
  float* Rf  = (float*)(ws + OFF_RF);   float* Kf  = (float*)(ws + OFF_KF);
  float* Vf  = (float*)(ws + OFF_VF);   float* Wd  = (float*)(ws + OFF_WD);
  float* Ab  = (float*)(ws + OFF_AB);   float* Vlr = (float*)(ws + OFF_VLR);
  float* KKf = (float*)(ws + OFF_KK);   float* Gf  = (float*)(ws + OFF_GF);
  float* Yf  = (float*)(ws + OFF_YF);

  auto tconv = [&](const float* s, __bf16* d, int K, int N) {
    conv_t_kernel<<<dim3(N / 32, K / 32), 256, 0, stream>>>(s, d, K, N);
  };
  tconv(W_r, WTR, CC, CC); tconv(W_k, WTK, CC, CC);
  tconv(W_v, WTV, CC, CC); tconv(W_o, WTO, CC, CC);
  tconv(w1, W1T, CC, 64);  tconv(w2, W2T, 64, CC);
  tconv(a1, A1T, CC, 64);  tconv(a2, A2T, 64, CC);
  tconv(v1, V1T, CC, 32);  tconv(v2, V2T, 32, CC);
  tconv(g1, G1T, CC, 128); tconv(g2, G2T, 128, CC);

  mix_kernel<<<(unsigned)(MC / 256), 256, 0, stream>>>(
      x, x_r, x_w, x_k, x_v, x_a, x_g, XR, XW, XK, XV, XA, XG);

  auto gemm = [&](const __bf16* A, const __bf16* Bt, float* Cf, __bf16* Cb,
                  int M, int N, int K, int ep) {
    dim3 grid((M + 127) / 128, (N + 127) / 128);
    gemm_bf16_wmma<<<grid, 128, 0, stream>>>(A, Bt, Cf, Cb, N, K, ep);
  };

  // big projections
  gemm(XR, WTR, Rf, nullptr, MROWS, CC, CC, 0);
  gemm(XK, WTK, Kf, nullptr, MROWS, CC, CC, 0);
  gemm(XV, WTV, Vf, nullptr, MROWS, CC, CC, 0);
  // low-rank paths
  gemm(XW, W1T, nullptr, HW, MROWS, 64, CC, 1);   // tanh
  gemm(HW, W2T, Wd, nullptr, MROWS, CC, 64, 0);
  gemm(XA, A1T, nullptr, HA, MROWS, 64, CC, 3);   // plain
  gemm(HA, A2T, Ab, nullptr, MROWS, CC, 64, 0);
  gemm(XV, V1T, nullptr, HV, MROWS, 32, CC, 3);   // plain
  gemm(HV, V2T, Vlr, nullptr, MROWS, CC, 32, 0);
  gemm(XG, G1T, nullptr, HG, MROWS, 128, CC, 2);  // sigmoid
  gemm(HG, G2T, Gf, nullptr, MROWS, CC, 128, 0);

  e1_kernel<<<dim3(MROWS, HH), 32, 0, stream>>>(
      Kf, Vf, Wd, Ab, Vlr, vfirst, w0, a0, v0, k_k, k_a, KKf);

  rwkv_rec_kernel<<<BB * HH, 256, 0, stream>>>(Rf, Wd, Kf, Vf, KKf, Ab, Yf);

  e2_kernel<<<dim3(MROWS, HH), 32, 0, stream>>>(
      Yf, Rf, Kf, Vf, Gf, r_k, ln_w, ln_b, YG);

  // final projection straight into d_out (first MC floats)
  gemm(YG, WTO, (float*)d_out, nullptr, MROWS, CC, CC, 0);

  // second tuple element: v_first passthrough
  copy4_f32_kernel<<<(unsigned)((MC / 4 + 255) / 256), 256, 0, stream>>>(
      vfirst, (float*)d_out + MC, MC / 4);
}
